// Mamba2Mixer_61211873903314
// MI455X (gfx1250) — compile-verified, matrix operand order checked by reading, not run on hardware
//
#include <hip/hip_runtime.h>
#include <hip/hip_bf16.h>
#include <stdint.h>

// ---------------------------------------------------------------- constants
#define T_SEQ     4096
#define DMODEL    2048
#define NHEADS_   64
#define HDIM      64
#define DSTATE    128
#define DINNER    4096
#define CONVDIM   4352
#define NPROJ     8512
#define NPROJ_PAD 8576   // 67 * 128
#define DCONV     4
#define CHUNK     256
#define NCHUNK    16     // T_SEQ / CHUNK

typedef __attribute__((ext_vector_type(16))) __bf16 v16bf;
typedef __attribute__((ext_vector_type(8)))  float  v8f;

// ---------------------------------------------------------------- GEMM (bf16 WMMA)
#define BM 128
#define BN 128
#define BK 32
#define LDSS 40   // LDS row stride in bf16 elems (80B, 16B aligned, conflict-skewed)

union FragU { uint32_t u[8]; v16bf v; uint4 q[2]; };

// Gather one 16x32 bf16 A fragment from LDS (row-major, K-contiguous rows)
// per the CDNA5 ISA layout: lane = 16*half + m; VGPR v<4: K = half*8 + 2v(+1),
// VGPR v>=4: K = 16 + half*8 + 2(v-4)(+1).
static __device__ __forceinline__ v16bf frag_from_lds(const __bf16* rowbase, int lh) {
  const uint32_t* p = (const uint32_t*)rowbase;   // row is 4B aligned (80B stride)
  FragU f;
  const int o = lh * 4;                            // u32 index = K/2
  f.u[0] = p[o + 0];  f.u[1] = p[o + 1];  f.u[2] = p[o + 2];  f.u[3] = p[o + 3];
  f.u[4] = p[o + 8];  f.u[5] = p[o + 9];  f.u[6] = p[o + 10]; f.u[7] = p[o + 11];
  return f.v;
}

// B fragment straight from an N-major (pre-transposed) global matrix:
// rowk0 = Bt + N_row*K + k0 + lh*8  -> two aligned 16B chunks.
static __device__ __forceinline__ v16bf frag_from_global(const __bf16* rowk0) {
  FragU f;
  f.q[0] = *(const uint4*)(rowk0);
  f.q[1] = *(const uint4*)(rowk0 + 16);
  return f.v;
}

// C[M][N] = A[M][K] (row-major bf16) x Bt[N][K] (N-major bf16), fp32 out.
// Double-buffered sA + register-pipelined B fragments: loads for step s+1
// overlap the 8 WMMAs of step s; one barrier per K-step.
__global__ __launch_bounds__(256, 2)
void gemm_bf16_wmma(const __bf16* __restrict__ A, const __bf16* __restrict__ Bt,
                    float* __restrict__ C, int M, int N, int K) {
  __shared__ alignas(16) __bf16 sA[2][BM * LDSS];

  const int tid  = threadIdx.x;
  const int lane = tid & 31;
  const int wave = tid >> 5;
  const int wm   = wave >> 2;      // 0..1 -> 64 rows each
  const int wn   = wave & 3;       // 0..3 -> 32 cols each
  const int lm   = lane & 15;
  const int lh   = lane >> 4;
  const int brow = blockIdx.y * BM;
  const int bcol = blockIdx.x * BN;

  // A tile loader: thread -> (row, 16-elem segment)
  const int arow = tid >> 1;
  const int aseg = (tid & 1) * 16;
  const __bf16* gA = A + (size_t)(brow + arow) * K + aseg;

  // B fragment base pointers
  const __bf16* pB0 = Bt + (size_t)(bcol + wn * 32 + lm) * K + lh * 8;
  const __bf16* pB1 = Bt + (size_t)(bcol + wn * 32 + 16 + lm) * K + lh * 8;

  const int aoff0 = (wm * 64 + 0 * 16 + lm) * LDSS;
  const int aoff1 = (wm * 64 + 1 * 16 + lm) * LDSS;
  const int aoff2 = (wm * 64 + 2 * 16 + lm) * LDSS;
  const int aoff3 = (wm * 64 + 3 * 16 + lm) * LDSS;

  v8f acc[4][2] = {};

  // ---- prologue: stage step 0
  {
    uint4 a0 = *(const uint4*)(gA);
    uint4 a1 = *(const uint4*)(gA + 8);
    *(uint4*)(&sA[0][arow * LDSS + aseg])     = a0;
    *(uint4*)(&sA[0][arow * LDSS + aseg + 8]) = a1;
  }
  v16bf b0 = frag_from_global(pB0);
  v16bf b1 = frag_from_global(pB1);
  __syncthreads();

  const int nsteps = K / BK;
  for (int s = 0; s < nsteps; s++) {
    const int buf = s & 1;
    const __bf16* sAb = sA[buf];
    const bool has_next = (s + 1 < nsteps);

    // issue next-step global loads first (overlap with WMMAs below)
    uint4 a0n, a1n;
    v16bf b0n, b1n;
    if (has_next) {
      const __bf16* gAn = gA + BK;
      a0n = *(const uint4*)(gAn);
      a1n = *(const uint4*)(gAn + 8);
      b0n = frag_from_global(pB0 + (size_t)(s + 1) * BK);
      b1n = frag_from_global(pB1 + (size_t)(s + 1) * BK);
      __builtin_prefetch(gAn + BK, 0, 1);                       // global_prefetch_b8
      __builtin_prefetch(pB0 + (size_t)(s + 2) * BK, 0, 1);
      __builtin_prefetch(pB1 + (size_t)(s + 2) * BK, 0, 1);
    }

    v16bf afr[4];
    afr[0] = frag_from_lds(sAb + aoff0, lh);
    afr[1] = frag_from_lds(sAb + aoff1, lh);
    afr[2] = frag_from_lds(sAb + aoff2, lh);
    afr[3] = frag_from_lds(sAb + aoff3, lh);

#pragma unroll
    for (int mt = 0; mt < 4; mt++) {
      acc[mt][0] = __builtin_amdgcn_wmma_f32_16x16x32_bf16(
          false, afr[mt], false, b0, (short)0, acc[mt][0], false, false);
      acc[mt][1] = __builtin_amdgcn_wmma_f32_16x16x32_bf16(
          false, afr[mt], false, b1, (short)0, acc[mt][1], false, false);
    }

    if (has_next) {
      __bf16* dst = (__bf16*)&sA[buf ^ 1][arow * LDSS + aseg];
      *(uint4*)(dst)     = a0n;
      *(uint4*)(dst + 8) = a1n;
      b0 = b0n;
      b1 = b1n;
      gA += BK;
    }
    __syncthreads();
  }

  // ---- store: C/D layout VGPR r, lane: M = r + 8*half, N = lane&15
#pragma unroll
  for (int mt = 0; mt < 4; mt++) {
#pragma unroll
    for (int nt = 0; nt < 2; nt++) {
      const int row0 = brow + wm * 64 + mt * 16 + lh * 8;
      const int col  = bcol + wn * 32 + nt * 16 + lm;
      float* cp = C + (size_t)row0 * N + col;
#pragma unroll
      for (int r = 0; r < 8; r++) cp[(size_t)r * N] = acc[mt][nt][r];
    }
  }
}

// ---------------------------------------------------------------- conversions
__global__ void cvt_f32_bf16(const float* __restrict__ s, __bf16* __restrict__ d, int n) {
  int i = blockIdx.x * 256 + threadIdx.x;
  if (i < n) d[i] = (__bf16)s[i];
}

// Wt[n][k] = bf16(W[k][n]); n in [0,Npad), zero pad for n >= N. 32x32 LDS tiles.
__global__ __launch_bounds__(256)
void transpose_f32_bf16(const float* __restrict__ W, __bf16* __restrict__ Wt,
                        int K, int N, int Npad) {
  __shared__ float tile[32][33];
  const int kb = blockIdx.x * 32;
  const int nb = blockIdx.y * 32;
  const int tx = threadIdx.x & 31;
  const int ty = threadIdx.x >> 5;   // 0..7
#pragma unroll
  for (int i = 0; i < 4; i++) {
    int k = kb + ty + 8 * i;
    int n = nb + tx;
    tile[ty + 8 * i][tx] = (n < N) ? W[(size_t)k * N + n] : 0.f;
  }
  __syncthreads();
#pragma unroll
  for (int i = 0; i < 4; i++) {
    int n = nb + ty + 8 * i;
    Wt[(size_t)n * K + kb + tx] = (__bf16)tile[tx][ty + 8 * i];
  }
}

// ---------------------------------------------------------------- causal conv1d + SiLU
__global__ void conv1d_silu(const float* __restrict__ zx, const float* __restrict__ w,
                            const float* __restrict__ b, float* __restrict__ out) {
  int i = blockIdx.x * 256 + threadIdx.x;
  if (i >= T_SEQ * CONVDIM) return;
  int t = i / CONVDIM, c = i % CONVDIM;
  float acc = b[c];
#pragma unroll
  for (int k = 0; k < DCONV; k++) {
    int tt = t - (DCONV - 1) + k;
    if (tt >= 0)
      acc = fmaf(zx[(size_t)tt * NPROJ_PAD + DINNER + c], w[c * DCONV + k], acc);
  }
  out[i] = acc / (1.0f + __expf(-acc));
}

// ---------------------------------------------------------------- dt softplus / dA
__global__ void dt_kernel(const float* __restrict__ zx, const float* __restrict__ dt_bias,
                          const float* __restrict__ Aarr, float* __restrict__ dt_act,
                          float* __restrict__ dAo) {
  int i = blockIdx.x * 256 + threadIdx.x;
  if (i >= T_SEQ * NHEADS_) return;
  int t = i / NHEADS_, h = i % NHEADS_;
  float x  = zx[(size_t)t * NPROJ_PAD + DINNER + CONVDIM + h] + dt_bias[h];
  float sp = (x > 20.f) ? x : log1pf(__expf(x));
  dt_act[i] = sp;
  dAo[i]    = __expf(sp * Aarr[h]);
}

// ---------------------------------------------------------------- chunk decay prefix
// Pp[t][h] = prod_{s=chunk_start..t} dA[s][h]; Ptot[c][h] = full-chunk product.
__global__ void chunk_decay(const float* __restrict__ dAv, float* __restrict__ Pp,
                            float* __restrict__ Ptot) {
  int i = blockIdx.x * 256 + threadIdx.x;
  if (i >= NHEADS_ * NCHUNK) return;
  int h = i % NHEADS_, c = i / NHEADS_;
  float prod = 1.f;
  for (int s = 0; s < CHUNK; s++) {
    int t = c * CHUNK + s;
    prod *= dAv[(size_t)t * NHEADS_ + h];
    Pp[(size_t)t * NHEADS_ + h] = prod;
  }
  Ptot[c * NHEADS_ + h] = prod;
}

// ---------------------------------------------------------------- pass 1: local scan
#define TCH 16
__global__ __launch_bounds__(256)
void scan_local(const float* __restrict__ xbc, const float* __restrict__ dt_act,
                const float* __restrict__ dAv, const float* __restrict__ Dv,
                float* __restrict__ y, float* __restrict__ Hpart) {
  __shared__ float sx[TCH * HDIM];
  __shared__ float sB[TCH * DSTATE];
  __shared__ float sC[TCH * DSTATE];
  __shared__ float sdt[TCH];
  __shared__ float sda[TCH];

  const int h   = blockIdx.x;      // head
  const int c   = blockIdx.y;      // time chunk
  const int tid = threadIdx.x;
  const int p   = tid >> 2;        // head-dim index 0..63
  const int q   = tid & 3;         // 4 lanes split D_STATE
  const int n0  = q * 32;
  const float Dh = Dv[h];
  const int tbase = c * CHUNK;

  float2 hreg[16];
#pragma unroll
  for (int j = 0; j < 16; j++) hreg[j] = make_float2(0.f, 0.f);

  for (int t0 = tbase; t0 < tbase + CHUNK; t0 += TCH) {
    for (int i = tid; i < TCH * HDIM; i += 256) {
      int s = i >> 6, j = i & 63;
      sx[i] = xbc[(size_t)(t0 + s) * CONVDIM + h * HDIM + j];
    }
    for (int i = tid; i < TCH * DSTATE; i += 256) {
      int s = i >> 7, j = i & 127;
      size_t rb = (size_t)(t0 + s) * CONVDIM + DINNER;
      sB[i] = xbc[rb + j];
      sC[i] = xbc[rb + DSTATE + j];
    }
    if (tid < TCH) {
      sdt[tid] = dt_act[(size_t)(t0 + tid) * NHEADS_ + h];
      sda[tid] = dAv[(size_t)(t0 + tid) * NHEADS_ + h];
    }
    __syncthreads();

#pragma unroll 1
    for (int s = 0; s < TCH; s++) {
      float xp   = sx[s * HDIM + p];
      float da   = sda[s];
      float coef = sdt[s] * xp;
      const float2* Bp = (const float2*)&sB[s * DSTATE + n0];
      const float2* Cp = (const float2*)&sC[s * DSTATE + n0];
      float ax = 0.f, ay = 0.f;
#pragma unroll
      for (int j = 0; j < 16; j++) {
        float2 b = Bp[j];
        float2 cc = Cp[j];
        hreg[j].x = fmaf(hreg[j].x, da, coef * b.x);
        hreg[j].y = fmaf(hreg[j].y, da, coef * b.y);
        ax = fmaf(hreg[j].x, cc.x, ax);
        ay = fmaf(hreg[j].y, cc.y, ay);
      }
      float acc = ax + ay;
      acc += __shfl_xor(acc, 1, 32);
      acc += __shfl_xor(acc, 2, 32);
      if (q == 0)
        y[(size_t)(t0 + s) * DINNER + h * HDIM + p] = fmaf(Dh, xp, acc);
    }
    __syncthreads();
  }

  // chunk-final local state
  float2* Hp = (float2*)(Hpart + ((size_t)c * NHEADS_ + h) * (HDIM * DSTATE)
                               + p * DSTATE + n0);
#pragma unroll
  for (int j = 0; j < 16; j++) Hp[j] = hreg[j];
}

// ---------------------------------------------------------------- pass 2: combine
// Hinit[c] = state entering chunk c; Hinit[0]=0, Hinit[c]=Ptot[c-1]*Hinit[c-1]+Hpart[c-1].
__global__ void chunk_combine(const float* __restrict__ Hpart, const float* __restrict__ Ptot,
                              float* __restrict__ Hinit) {
  const int i = blockIdx.x * 256 + threadIdx.x;    // over NHEADS_*HDIM*DSTATE
  if (i >= NHEADS_ * HDIM * DSTATE) return;
  const int h = i / (HDIM * DSTATE);
  const size_t stride = (size_t)NHEADS_ * HDIM * DSTATE;
  float cur = 0.f;
  for (int c = 0; c < NCHUNK; c++) {
    Hinit[(size_t)c * stride + i] = cur;
    cur = fmaf(cur, Ptot[c * NHEADS_ + h], Hpart[(size_t)c * stride + i]);
  }
}

// ---------------------------------------------------------------- pass 3: correction
// y[t] += Pp[t] * (C_t . Hinit[chunk(t)]), Hinit tile resident in LDS (32KB).
#define CST 16
__global__ __launch_bounds__(256)
void chunk_correct(const float* __restrict__ Hinit, const float* __restrict__ xbc,
                   const float* __restrict__ Pp, float* __restrict__ y) {
  __shared__ float sH[HDIM * DSTATE];     // 32KB
  __shared__ float sCc[CST * DSTATE];
  __shared__ float sPp[CST];
  const int h   = blockIdx.x;
  const int c   = blockIdx.y;
  const int tid = threadIdx.x;
  const int p   = tid >> 2;
  const int q   = tid & 3;
  const int n0  = q * 32;

  const size_t hb = ((size_t)c * NHEADS_ + h) * (HDIM * DSTATE);
  for (int i = tid; i < HDIM * DSTATE; i += 256) sH[i] = Hinit[hb + i];
  const float2* Hp = (const float2*)&sH[p * DSTATE + n0];

  for (int s0 = 0; s0 < CHUNK; s0 += CST) {
    __syncthreads();
    for (int i = tid; i < CST * DSTATE; i += 256) {
      int s = i >> 7, j = i & 127;
      sCc[i] = xbc[(size_t)(c * CHUNK + s0 + s) * CONVDIM + DINNER + DSTATE + j];
    }
    if (tid < CST)
      sPp[tid] = Pp[(size_t)(c * CHUNK + s0 + tid) * NHEADS_ + h];
    __syncthreads();

#pragma unroll 1
    for (int s = 0; s < CST; s++) {
      const float2* Cp = (const float2*)&sCc[s * DSTATE + n0];
      float ax = 0.f, ay = 0.f;
#pragma unroll
      for (int j = 0; j < 16; j++) {
        float2 hv = Hp[j];
        float2 cv = Cp[j];
        ax = fmaf(hv.x, cv.x, ax);
        ay = fmaf(hv.y, cv.y, ay);
      }
      float acc = ax + ay;
      acc += __shfl_xor(acc, 1, 32);
      acc += __shfl_xor(acc, 2, 32);
      if (q == 0) {
        size_t yi = (size_t)(c * CHUNK + s0 + s) * DINNER + h * HDIM + p;
        y[yi] += sPp[s] * acc;
      }
    }
  }
}

// ---------------------------------------------------------------- gate + RMSNorm -> bf16
__global__ __launch_bounds__(256)
void gate_rmsnorm_bf16(const float* __restrict__ y, const float* __restrict__ zx,
                       const float* __restrict__ norm_w, __bf16* __restrict__ yb) {
  __shared__ float red[8];
  const int t   = blockIdx.x;
  const int tid = threadIdx.x;
  float g[16];
  float ss = 0.f;
#pragma unroll
  for (int j = 0; j < 16; j++) {
    int c   = tid + j * 256;
    float z = zx[(size_t)t * NPROJ_PAD + c];
    float v = y[(size_t)t * DINNER + c] * (z / (1.f + __expf(-z)));
    g[j] = v;
    ss   = fmaf(v, v, ss);
  }
#pragma unroll
  for (int o = 16; o >= 1; o >>= 1) ss += __shfl_xor(ss, o, 32);
  if ((tid & 31) == 0) red[tid >> 5] = ss;
  __syncthreads();
  float tot = red[0] + red[1] + red[2] + red[3] + red[4] + red[5] + red[6] + red[7];
  float scale = rsqrtf(tot / (float)DINNER + 1e-5f);
#pragma unroll
  for (int j = 0; j < 16; j++) {
    int c = tid + j * 256;
    yb[(size_t)t * DINNER + c] = (__bf16)(g[j] * scale * norm_w[c]);
  }
}

// ---------------------------------------------------------------- launch
extern "C" void kernel_launch(void* const* d_in, const int* in_sizes, int n_in,
                              void* d_out, int out_size, void* d_ws, size_t ws_size,
                              hipStream_t stream) {
  const float* hidden  = (const float*)d_in[0];
  const float* W_in    = (const float*)d_in[1];
  const float* conv_w  = (const float*)d_in[2];
  const float* conv_b  = (const float*)d_in[3];
  const float* Aarr    = (const float*)d_in[4];
  const float* Dvals   = (const float*)d_in[5];
  const float* dt_bias = (const float*)d_in[6];
  const float* norm_w  = (const float*)d_in[7];
  const float* W_out   = (const float*)d_in[8];
  float* out = (float*)d_out;

  char* ws = (char*)d_ws;
  size_t off = 0;
  auto alloc = [&](size_t bytes) -> char* {
    char* p = ws + off;
    off += (bytes + 255) & ~(size_t)255;
    return p;
  };
  __bf16* hid_bf   = (__bf16*)alloc((size_t)T_SEQ * DMODEL * 2);
  __bf16* win_bfT  = (__bf16*)alloc((size_t)NPROJ_PAD * DMODEL * 2);  // [N][K]
  float*  zx       = (float*) alloc((size_t)T_SEQ * NPROJ_PAD * 4);
  float*  xbc      = (float*) alloc((size_t)T_SEQ * CONVDIM * 4);
  float*  dt_act   = (float*) alloc((size_t)T_SEQ * NHEADS_ * 4);
  float*  dAbuf    = (float*) alloc((size_t)T_SEQ * NHEADS_ * 4);
  float*  ybuf     = (float*) alloc((size_t)T_SEQ * DINNER * 4);
  __bf16* y_bf     = (__bf16*)alloc((size_t)T_SEQ * DINNER * 2);
  __bf16* wout_bfT = (__bf16*)alloc((size_t)DMODEL * DINNER * 2);     // [N][K]
  float*  Pp       = (float*) alloc((size_t)T_SEQ * NHEADS_ * 4);
  float*  Ptot     = (float*) alloc((size_t)NCHUNK * NHEADS_ * 4);
  float*  Hpart    = (float*) alloc((size_t)NCHUNK * NHEADS_ * HDIM * DSTATE * 4);
  float*  Hinit    = (float*) alloc((size_t)NCHUNK * NHEADS_ * HDIM * DSTATE * 4);

  int n1 = T_SEQ * DMODEL;
  cvt_f32_bf16<<<(n1 + 255) / 256, 256, 0, stream>>>(hidden, hid_bf, n1);

  {   // W_in: (K=2048) x (N=8512) -> [Npad=8576][2048]
    dim3 g(DMODEL / 32, NPROJ_PAD / 32);
    transpose_f32_bf16<<<g, 256, 0, stream>>>(W_in, win_bfT, DMODEL, NPROJ, NPROJ_PAD);
  }
  {   // W_out: (K=4096) x (N=2048) -> [2048][4096]
    dim3 g(DINNER / 32, DMODEL / 32);
    transpose_f32_bf16<<<g, 256, 0, stream>>>(W_out, wout_bfT, DINNER, DMODEL, DMODEL);
  }

  dim3 g1(NPROJ_PAD / BN, T_SEQ / BM);
  gemm_bf16_wmma<<<g1, 256, 0, stream>>>(hid_bf, win_bfT, zx, T_SEQ, NPROJ_PAD, DMODEL);

  int nc = T_SEQ * CONVDIM;
  conv1d_silu<<<(nc + 255) / 256, 256, 0, stream>>>(zx, conv_w, conv_b, xbc);
  int nd = T_SEQ * NHEADS_;
  dt_kernel<<<(nd + 255) / 256, 256, 0, stream>>>(zx, dt_bias, Aarr, dt_act, dAbuf);

  chunk_decay<<<(NHEADS_ * NCHUNK + 255) / 256, 256, 0, stream>>>(dAbuf, Pp, Ptot);

  dim3 gs(NHEADS_, NCHUNK);
  scan_local<<<gs, 256, 0, stream>>>(xbc, dt_act, dAbuf, Dvals, ybuf, Hpart);

  int nstate = NHEADS_ * HDIM * DSTATE;
  chunk_combine<<<(nstate + 255) / 256, 256, 0, stream>>>(Hpart, Ptot, Hinit);

  chunk_correct<<<gs, 256, 0, stream>>>(Hinit, xbc, Pp, ybuf);

  gate_rmsnorm_bf16<<<T_SEQ, 256, 0, stream>>>(ybuf, zx, norm_w, y_bf);

  dim3 g2(DMODEL / BN, T_SEQ / BM);
  gemm_bf16_wmma<<<g2, 256, 0, stream>>>(y_bf, wout_bfT, out, T_SEQ, DMODEL, DINNER);
}
